// GraphConvUNet_41815801594441
// MI455X (gfx1250) — matrix operand order repaired and checked by reading, not compiled, not run on hardware
//
#include <hip/hip_runtime.h>

typedef __attribute__((ext_vector_type(16))) __bf16 v16bf;
typedef __attribute__((ext_vector_type(8)))  float  v8f;

#define V_NODES 40000
#define N_EDGES 320000
#define NLAYERS 17

// ---------------- device helpers ----------------

__device__ __forceinline__ unsigned short f32_to_bf16(float f) {
    unsigned u = __float_as_uint(f);
    unsigned r = u + 0x7FFFu + ((u >> 16) & 1u);   // round-to-nearest-even
    return (unsigned short)(r >> 16);
}

// ---------------- elementwise kernels ----------------

__global__ void k_zero(float* p, int n) {
    int t = blockIdx.x * blockDim.x + threadIdx.x;
    if (t < n) p[t] = 0.f;
}

__global__ void k_copy(const float* s, float* d, int n) {
    int t = blockIdx.x * blockDim.x + threadIdx.x;
    if (t < n) d[t] = s[t];
}

// z = 2*z - x0   (Chebyshev recurrence finalize)
__global__ void k_axpby(float* z, const float* x0, int n) {
    int t = blockIdx.x * blockDim.x + threadIdx.x;
    if (t < n) z[t] = 2.0f * z[t] - x0[t];
}

// transpose-convert weights: w[kk, o] (f32) -> wbT[o, kk] (bf16 bits)
__global__ void k_w2bf_t(const float* w, unsigned short* o, int Kdim, int coutShift, int n) {
    int t = blockIdx.x * blockDim.x + threadIdx.x;
    if (t >= n) return;
    int kk = t >> coutShift;
    int c  = t & ((1 << coutShift) - 1);
    o[(size_t)c * Kdim + kk] = f32_to_bf16(w[t]);
}

// copy skip channels into concat buffer columns [colOff, colOff+C)
__global__ void k_copy_cols(const float* src, float* dst, int ldDst, int colOff,
                            int cshift, int n) {
    int t = blockIdx.x * blockDim.x + threadIdx.x;
    if (t >= n) return;
    int v = t >> cshift;
    int c = t & ((1 << cshift) - 1);
    dst[(size_t)v * ldDst + colOff + c] = src[t];
}

// ---------------- SpMM: z += L * y   (COO, edge-parallel, 4 ch/thread) ----------------

__global__ void k_spmm(const int* __restrict__ rows, const int* __restrict__ cols,
                       const float* __restrict__ vals,
                       const float* __restrict__ y, float* __restrict__ z,
                       int cshift /* log2(C), C >= 4 */) {
    long t = (long)blockIdx.x * blockDim.x + threadIdx.x;
    int qshift = cshift - 2;                 // quads per row
    long e = t >> qshift;
    if (e >= N_EDGES) return;
    int c4 = (int)(t & ((1 << qshift) - 1)) << 2;

    float v  = vals[e];
    long  cl = cols[e];
    long  rw = rows[e];

    const float4 yv = *(const float4*)(y + (cl << cshift) + c4);
    float* zp = z + (rw << cshift) + c4;
    unsafeAtomicAdd(zp + 0, v * yv.x);
    unsafeAtomicAdd(zp + 1, v * yv.y);
    unsafeAtomicAdd(zp + 2, v * yv.z);
    unsafeAtomicAdd(zp + 3, v * yv.w);
}

// ---------------- Chebyshev GEMM ----------------
// out[v,o] = sum_kk A[v,kk]*W[kk,o] + bias,  A[v,kk] = xk[kk>>cs][v][kk&(cin-1)]
// Block: 256 thr = 8 waves as 4M x 2N; block tile 128 rows x 64 cols.
// Wave: 2x2 register tile of 16x16 WMMA accumulators (4 wmma / K-chunk).
// LDS tiles stored in fragment order -> ds_load_b128 fragment builds.
//  A frag (16-bit 16x32): lane hi=0 holds K{0..7,16..23}, hi=1 holds K{8..15,24..31}
//    row-local position p(kloc) = ((kloc>>3)&1)*16 + (kloc&7) + (kloc>=16 ? 8 : 0)
//  B frag (16-bit 32x16): same k order as A (hi=0 K{0..7,16..23}, hi=1 K{8..15,24..31}) -> same p(kloc).
// B tile staged via GLOBAL_LOAD_ASYNC_TO_LDS_B128 (ASYNCcnt), 16B/lane.

#define A_STRIDE 40   // ushorts (80B): 16 distinct banks across 16 lanes, 16B aligned
#define B_STRIDE 40

template <bool GENERIC>
__global__ __launch_bounds__(256)
void k_cheb_gemm(const float* __restrict__ x0, const float* __restrict__ x1,
                 const float* __restrict__ x2,
                 const unsigned short* __restrict__ wbT, const float* __restrict__ bias,
                 int cinShift, int cout,
                 float* __restrict__ out, int ldOut, int colOff,
                 const float* __restrict__ skipAdd, int act) {
    const int cin  = 1 << cinShift;
    const int Kdim = 3 * cin;

    __shared__ __align__(16) unsigned short lA[128 * A_STRIDE];
    __shared__ __align__(16) unsigned short lB[64 * B_STRIDE];

    const int tid  = threadIdx.x;
    const int wave = tid >> 5;
    const int lane = tid & 31;
    const int wm   = wave >> 1;       // 0..3
    const int wn   = wave & 1;        // 0..1
    const int m    = lane & 15;
    const int hi   = lane >> 4;
    const int rowBase = blockIdx.x * 128;
    const int oBase   = blockIdx.y * 64;

    // staging assignments (fixed per thread)
    const int rA    = tid >> 1;              // 0..127
    const int kbA   = (tid & 1) << 4;        // 0 or 16
    const int vrowA = rowBase + rA;
    const int nB    = tid & 63;              // 0..63
    const int kbB   = (tid >> 6) << 3;       // 0,8,16,24
    const int oB    = oBase + nB;

    // per-thread fixed LDS slots
    unsigned short* aSlot = &lA[rA * A_STRIDE + (kbA ? 8 : 0)];
    unsigned short* bSlot = &lB[nB * B_STRIDE + (((kbB & 8) << 1) | ((kbB & 16) >> 1))];
    const unsigned bLdsOff = (unsigned)(unsigned long long)bSlot;  // LDS byte offset (aperture low bits)
    const unsigned short* bGlob = wbT + (size_t)oB * Kdim + kbB;

    v8f acc00 = {}, acc01 = {}, acc10 = {}, acc11 = {};

    for (int kk0 = 0; kk0 < Kdim; kk0 += 32) {
        // ---- stage A: this thread covers kloc [kbA, kbA+16)
        {
            unsigned short tmp[16];
            if (!GENERIC) {
                if (vrowA < V_NODES) {
                    // whole chunk inside one Chebyshev segment (cin >= 32)
                    int kk = kk0 + kbA;
                    const float* src = (kk < cin) ? x0 : ((kk < 2 * cin) ? x1 : x2);
                    const float* p = src + ((size_t)vrowA << cinShift) + (kk & (cin - 1));
                    float4 f0 = ((const float4*)p)[0];
                    float4 f1 = ((const float4*)p)[1];
                    float4 f2 = ((const float4*)p)[2];
                    float4 f3 = ((const float4*)p)[3];
                    __builtin_prefetch(p + 32, 0, 0);   // next K-chunk -> global_prefetch
                    float fv[16] = {f0.x,f0.y,f0.z,f0.w, f1.x,f1.y,f1.z,f1.w,
                                    f2.x,f2.y,f2.z,f2.w, f3.x,f3.y,f3.z,f3.w};
#pragma unroll
                    for (int j = 0; j < 16; ++j) tmp[j] = f32_to_bf16(fv[j]);
                } else {
#pragma unroll
                    for (int j = 0; j < 16; ++j) tmp[j] = 0;
                }
            } else {
#pragma unroll
                for (int j = 0; j < 16; ++j) {
                    int kk = kk0 + kbA + j;
                    float val = 0.f;
                    if (vrowA < V_NODES && kk < Kdim) {
                        int ksel = kk >> cinShift;
                        int i    = kk & (cin - 1);
                        const float* src = (ksel == 0) ? x0 : ((ksel == 1) ? x1 : x2);
                        val = src[((size_t)vrowA << cinShift) + i];
                    }
                    tmp[j] = f32_to_bf16(val);
                }
            }
            unsigned q[8];
#pragma unroll
            for (int j = 0; j < 8; ++j)
                q[j] = (unsigned)tmp[2 * j] | ((unsigned)tmp[2 * j + 1] << 16);
            uint4 s0 = {q[0], q[1], q[2], q[3]};
            uint4 s1 = {q[4], q[5], q[6], q[7]};
            *(uint4*)(aSlot)      = s0;                   // p-runs: {0,16} or {8,24}
            *(uint4*)(aSlot + 16) = s1;
        }
        // ---- stage B: kloc [kbB, kbB+8) of column nB, async DMA to LDS
        if (!GENERIC) {
            if (oB < cout) {
                const unsigned short* ga = bGlob + kk0;
                asm volatile("global_load_async_to_lds_b128 %0, %1, off"
                             :: "v"(bLdsOff), "v"(ga) : "memory");
            } else {
                uint4 zz = {0u, 0u, 0u, 0u};
                *(uint4*)bSlot = zz;
            }
        } else {
            unsigned short tb[8];
#pragma unroll
            for (int j = 0; j < 8; ++j) {
                int kk = kk0 + kbB + j;
                tb[j] = (oB < cout && kk < Kdim) ? wbT[(size_t)oB * Kdim + kk] : 0;
            }
            uint4 s;
            s.x = (unsigned)tb[0] | ((unsigned)tb[1] << 16);
            s.y = (unsigned)tb[2] | ((unsigned)tb[3] << 16);
            s.z = (unsigned)tb[4] | ((unsigned)tb[5] << 16);
            s.w = (unsigned)tb[6] | ((unsigned)tb[7] << 16);
            *(uint4*)bSlot = s;
        }
        if (!GENERIC) {
            asm volatile("s_wait_asynccnt 0x0" ::: "memory");
        }
        __syncthreads();

        // ---- fragments: 2x ds_load_b128 each
        union Frag { uint4 q[2]; v16bf v; };
        Frag a0, a1, b0, b1;
        {
            const int ar0 = (wm * 32 + m)      * A_STRIDE + hi * 16;
            const int ar1 = (wm * 32 + 16 + m) * A_STRIDE + hi * 16;
            a0.q[0] = *(const uint4*)&lA[ar0];
            a0.q[1] = *(const uint4*)&lA[ar0 + 8];
            a1.q[0] = *(const uint4*)&lA[ar1];
            a1.q[1] = *(const uint4*)&lA[ar1 + 8];
            const int bc0 = (wn * 32 + m)      * B_STRIDE + hi * 16;
            const int bc1 = (wn * 32 + 16 + m) * B_STRIDE + hi * 16;
            b0.q[0] = *(const uint4*)&lB[bc0];
            b0.q[1] = *(const uint4*)&lB[bc0 + 8];
            b1.q[0] = *(const uint4*)&lB[bc1];
            b1.q[1] = *(const uint4*)&lB[bc1 + 8];
        }

        acc00 = __builtin_amdgcn_wmma_f32_16x16x32_bf16(false, a0.v, false, b0.v, (short)0, acc00, false, false);
        acc01 = __builtin_amdgcn_wmma_f32_16x16x32_bf16(false, a0.v, false, b1.v, (short)0, acc01, false, false);
        acc10 = __builtin_amdgcn_wmma_f32_16x16x32_bf16(false, a1.v, false, b0.v, (short)0, acc10, false, false);
        acc11 = __builtin_amdgcn_wmma_f32_16x16x32_bf16(false, a1.v, false, b1.v, (short)0, acc11, false, false);
        __syncthreads();
    }

    // ---- epilogue: bias + leaky-relu + residual + concat offset
    auto store_tile = [&](const v8f& a, int tm, int tn) {
        int o = oBase + wn * 32 + tn * 16 + m;
        if (o >= cout) return;
        float bv = bias[o];
#pragma unroll
        for (int r = 0; r < 8; ++r) {
            int vrow = rowBase + wm * 32 + tm * 16 + r + (hi ? 8 : 0);
            if (vrow < V_NODES) {
                float val = a[r] + bv;
                if (act) val = (val > 0.f) ? val : 0.01f * val;
                if (skipAdd) val += skipAdd[(size_t)vrow * cout + o];
                out[(size_t)vrow * ldOut + colOff + o] = val;
            }
        }
    };
    store_tile(acc00, 0, 0);
    store_tile(acc01, 0, 1);
    store_tile(acc10, 1, 0);
    store_tile(acc11, 1, 1);
}

// ---------------- host orchestration ----------------

static const int CIN[NLAYERS]  = {16,64,64,64,64,128,128,128,128,256,256,256,256,512,512,256,128};
static const int COUT[NLAYERS] = {64,64,64,64,128,128,128,128,256,256,256,256,512,256,128,64,16};

extern "C" void kernel_launch(void* const* d_in, const int* in_sizes, int n_in,
                              void* d_out, int out_size, void* d_ws, size_t ws_size,
                              hipStream_t stream) {
    (void)in_sizes; (void)n_in; (void)out_size; (void)ws_size;

    const int*   rows = (const int*)d_in[0];
    const int*   cols = (const int*)d_in[1];
    const float* vals = (const float*)d_in[2];
    const float* x_in = (const float*)d_in[3];
    const float* Wt[NLAYERS];
    const float* Bs[NLAYERS];
    for (int i = 0; i < NLAYERS; ++i) {
        Wt[i] = (const float*)d_in[4 + i];
        Bs[i] = (const float*)d_in[4 + NLAYERS + i];
    }

    // workspace carve-out
    char* ws = (char*)d_ws;
    size_t off = 0;
    auto alloc = [&](size_t bytes) -> void* {
        void* p = ws + off;
        off += (bytes + 255) & ~(size_t)255;
        return p;
    };
    const size_t VS = (size_t)V_NODES;
    float* bufA = (float*)alloc(VS * 512 * 4);
    float* bufB = (float*)alloc(VS * 512 * 4);
    float* T1   = (float*)alloc(VS * 512 * 4);
    float* T2   = (float*)alloc(VS * 512 * 4);
    float* S0   = (float*)alloc(VS * 64  * 4);
    float* S1   = (float*)alloc(VS * 128 * 4);
    float* S2   = (float*)alloc(VS * 256 * 4);
    size_t woff[NLAYERS]; size_t wtot = 0;
    for (int i = 0; i < NLAYERS; ++i) { woff[i] = wtot; wtot += (size_t)3 * CIN[i] * COUT[i]; }
    unsigned short* WBT = (unsigned short*)alloc(wtot * 2);

    // transpose-convert all layer weights to bf16 [o, kk]
    for (int i = 0; i < NLAYERS; ++i) {
        int n = 3 * CIN[i] * COUT[i];
        k_w2bf_t<<<(n + 255) / 256, 256, 0, stream>>>(Wt[i], WBT + woff[i],
                                                      3 * CIN[i], __builtin_ctz(COUT[i]), n);
    }

    auto conv = [&](const float* X, int li, float* outp, int ldOut, int colOff,
                    const float* skipAdd, int act) {
        int cin = CIN[li], cout = COUT[li];
        int csh = __builtin_ctz(cin);
        int n   = V_NODES * cin;
        long sp = (long)N_EDGES * (cin >> 2);
        k_zero<<<(n + 255) / 256, 256, 0, stream>>>(T1, n);
        k_spmm<<<(unsigned)((sp + 255) / 256), 256, 0, stream>>>(rows, cols, vals, X, T1, csh);
        k_zero<<<(n + 255) / 256, 256, 0, stream>>>(T2, n);
        k_spmm<<<(unsigned)((sp + 255) / 256), 256, 0, stream>>>(rows, cols, vals, T1, T2, csh);
        k_axpby<<<(n + 255) / 256, 256, 0, stream>>>(T2, X, n);
        dim3 g((V_NODES + 127) / 128, (cout + 63) / 64);
        if (cin >= 32) {
            k_cheb_gemm<false><<<g, 256, 0, stream>>>(X, T1, T2, WBT + woff[li], Bs[li],
                                                      csh, cout, outp, ldOut, colOff, skipAdd, act);
        } else {
            k_cheb_gemm<true><<<g, 256, 0, stream>>>(X, T1, T2, WBT + woff[li], Bs[li],
                                                     csh, cout, outp, ldOut, colOff, skipAdd, act);
        }
    };

    // ---- layer 0
    conv(x_in, 0, bufA, 64, 0, nullptr, 1);
    const float* x = bufA;
    int C = 64, li = 1;
    float* Sk[3] = {S0, S1, S2};

    // ---- encoder: 3 stages of (3 convs + residual, then transition conv)
    for (int s = 0; s < 3; ++s) {
        int n = V_NODES * C;
        k_copy<<<(n + 255) / 256, 256, 0, stream>>>(x, Sk[s], n);
        for (int j = 0; j < 3; ++j) {
            float* dst = (x == bufA) ? bufB : bufA;
            conv(x, li, dst, C, 0, (j == 2) ? Sk[s] : nullptr, 1);
            x = dst; ++li;
        }
        float* dst = (x == bufA) ? bufB : bufA;
        conv(x, li, dst, COUT[li], 0, nullptr, 1);
        C = COUT[li]; x = dst; ++li;
    }

    // ---- decoder: conv then concat with reversed skips
    int skC[3] = {64, 128, 256};
    for (int d = 0; d < 3; ++d) {
        int sidx = 2 - d;
        int co = COUT[li];
        int Ct = co + skC[sidx];
        float* dst = (x == bufA) ? bufB : bufA;
        conv(x, li, dst, Ct, 0, nullptr, 1); ++li;
        int n = V_NODES * skC[sidx];
        k_copy_cols<<<(n + 255) / 256, 256, 0, stream>>>(Sk[sidx], dst, Ct, co,
                                                         __builtin_ctz(skC[sidx]), n);
        x = dst; C = Ct;
    }

    // ---- final projection, no activation, straight to d_out
    conv(x, 16, (float*)d_out, 16, 0, nullptr, 0);
}
